// PhysicsAugEGATModel_86088324481792
// MI455X (gfx1250) — compile-verified, matrix operand order checked by reading, not run on hardware
//
#include <hip/hip_runtime.h>
#include <cstdint>
#include <cstddef>

// ---------------------------------------------------------------------------
// MI455X / gfx1250 implementation of the physics-augmented EGAT model.
//  * All 128-wide linears run as v_wmma_f32_16x16x32_bf16 (f32 accumulate).
//  * Weight panels are pre-transposed to bf16 [N][K] and DMA'd into LDS with
//    the Tensor Data Mover (tensor_load_to_lds + s_wait_tensorcnt).
//  * Odd GNN layers skip attention entirely (reference discards nout there).
//  * All E x 128 streaming/gather/scatter kernels use 16B vector accesses.
// ---------------------------------------------------------------------------

typedef __attribute__((ext_vector_type(16))) __bf16 v16bf;
typedef __attribute__((ext_vector_type(8)))  float  v8f;
typedef __attribute__((ext_vector_type(4)))  unsigned int u32x4;
typedef __attribute__((ext_vector_type(8)))  int    i32x8;
typedef __attribute__((ext_vector_type(4)))  int    i32x4;

#define CEILDIV(a, b) (((a) + (b) - 1) / (b))

__device__ __forceinline__ unsigned short f2bf(float f) {
  unsigned u = __float_as_uint(f);
  unsigned r = (u + 0x7FFFu + ((u >> 16) & 1u)) >> 16;   // round-nearest-even
  return (unsigned short)r;
}
__device__ __forceinline__ float bf2f(unsigned short h) {
  return __uint_as_float(((unsigned)h) << 16);
}
// order-preserving float <-> uint encoding for atomicMax-based segment_max
__device__ __forceinline__ unsigned fenc(float f) {
  unsigned u = __float_as_uint(f);
  return (u & 0x80000000u) ? ~u : (u | 0x80000000u);
}
__device__ __forceinline__ float fdec(unsigned u) {
  return (u & 0x80000000u) ? __uint_as_float(u & 0x7FFFFFFFu)
                           : __uint_as_float(~u);
}

#if defined(__gfx1250__) && __has_builtin(__builtin_amdgcn_tensor_load_to_lds) && \
    __has_builtin(__builtin_amdgcn_s_wait_tensorcnt)
#define USE_TDM 1
#endif

// ---------------------------------------------------------------------------
// WMMA GEMM:  C[M x N] = act(A[M x K](bf16) * W[K x N] + bias)
// Wt is the weight pre-transposed to [N][K] bf16 so both A and B fragments
// load as two contiguous 16B chunks per lane (K halves at +0 and +16).
// Block = 256 threads = 8 waves; wave w owns rows [blk*128+16w, +16) and all
// N/16 column tiles (B panel reused from LDS).  WF/WB select f32/bf16 output
// at compile time so the epilogue is branch-free on the hot path.
// ---------------------------------------------------------------------------
template <int K, int N, int WF, int WB, bool RELU>
__global__ __launch_bounds__(256) void gemm_wmma(
    const unsigned short* __restrict__ A, const unsigned short* __restrict__ Wt,
    const float* __restrict__ bias, float* __restrict__ Cf,
    unsigned short* __restrict__ Cb, int M) {
  __shared__ unsigned short w_lds[N * K];
  const int tid = threadIdx.x;

#ifdef USE_TDM
  if ((tid >> 5) == 0) {  // wave 0 issues one TDM DMA for the whole panel
    unsigned lds_off = (unsigned)(size_t)(&w_lds[0]);
    unsigned long long ga = (unsigned long long)(size_t)Wt;
    u32x4 g0;
    g0[0] = 1u;                       // count=1 valid descriptor
    g0[1] = lds_off;                  // lds_addr
    g0[2] = (unsigned)ga;             // global_addr[31:0]
    g0[3] = ((unsigned)(ga >> 32) & 0x01FFFFFFu) | (2u << 30);  // addr hi | type=2
    i32x8 g1;
    g1[0] = (int)(1u << 16);                                   // data_size = 2B
    g1[1] = (int)(((unsigned)K & 0xFFFFu) << 16);              // tensor_dim0 lo16
    g1[2] = (int)((((unsigned)K >> 16) & 0xFFFFu) | (((unsigned)N & 0xFFFFu) << 16));
    g1[3] = (int)((((unsigned)N >> 16) & 0xFFFFu) | (((unsigned)K & 0xFFFFu) << 16)); // tile_dim0=K
    g1[4] = (int)((unsigned)N & 0xFFFFu);                      // tile_dim1=N, tile_dim2=0
    g1[5] = (int)K;                                            // dim0 stride lo32
    g1[6] = (int)(((unsigned)K & 0xFFFFu) << 16);              // dim1 stride lo16
    g1[7] = 0;
    i32x4 z4 = (i32x4)0;
#if __clang_major__ >= 23
    i32x8 z8 = (i32x8)0;
    __builtin_amdgcn_tensor_load_to_lds(g0, g1, z4, z4, z8, 0);
#else
    __builtin_amdgcn_tensor_load_to_lds(g0, g1, z4, z4, 0);
#endif
    __builtin_amdgcn_s_wait_tensorcnt(0);
  }
#else
  for (int i = tid * 8; i < N * K; i += 256 * 8)
    *(u32x4*)(&w_lds[i]) = *(const u32x4*)(Wt + i);
#endif
  __syncthreads();

  const int wave = tid >> 5, lane = tid & 31;
  const int lr = lane & 15;
  const int koff = (lane >> 4) << 3;  // lanes 16..31 hold K halves +8
  const int rowBase = blockIdx.x * 128 + wave * 16;
  int row = rowBase + lr;
  if (row >= M) row = M - 1;  // clamp: keeps EXEC all-1s for WMMA, stores masked
  const unsigned short* arow = A + (size_t)row * K;

  constexpr int NT = N / 16;
  v8f acc[NT];
#pragma unroll
  for (int t = 0; t < NT; ++t) acc[t] = (v8f)0.0f;

#pragma unroll
  for (int k0 = 0; k0 < K; k0 += 32) {
    union { u32x4 q[2]; v16bf v; } af;
    af.q[0] = *(const u32x4*)(arow + k0 + koff);
    af.q[1] = *(const u32x4*)(arow + k0 + koff + 16);
#pragma unroll
    for (int t = 0; t < NT; ++t) {
      const unsigned short* brow = &w_lds[(size_t)(t * 16 + lr) * K + k0 + koff];
      union { u32x4 q[2]; v16bf v; } bf_;
      bf_.q[0] = *(const u32x4*)(brow);
      bf_.q[1] = *(const u32x4*)(brow + 16);
      acc[t] = __builtin_amdgcn_wmma_f32_16x16x32_bf16(
          false, af.v, false, bf_.v, (short)0, acc[t], false, false);
    }
  }

  const int mBase = (lane >> 4) * 8;  // C layout: lanes 16..31 hold M rows +8
  const bool full = (rowBase + 16) <= M;
  if (full) {
#pragma unroll
    for (int t = 0; t < NT; ++t) {
      const int n = t * 16 + lr;
      const float bv = bias[n];
      size_t o = (size_t)(rowBase + mBase) * N + n;
#pragma unroll
      for (int r = 0; r < 8; ++r, o += N) {
        float v = acc[t][r] + bv;
        if (RELU) v = fmaxf(v, 0.0f);
        if (WF) Cf[o] = v;
        if (WB) Cb[o] = f2bf(v);
      }
    }
  } else {
#pragma unroll
    for (int t = 0; t < NT; ++t) {
      const int n = t * 16 + lr;
      const float bv = bias[n];
#pragma unroll
      for (int r = 0; r < 8; ++r) {
        int m = rowBase + mBase + r;
        if (m < M) {
          float v = acc[t][r] + bv;
          if (RELU) v = fmaxf(v, 0.0f);
          if (WF) Cf[(size_t)m * N + n] = v;
          if (WB) Cb[(size_t)m * N + n] = f2bf(v);
        }
      }
    }
  }
}

// ---------------------------------------------------------------------------
// Small/scalar kernels
// ---------------------------------------------------------------------------
__global__ void pack_wt(const float* __restrict__ w, unsigned short* __restrict__ wt,
                        int K, int N) {
  int i = blockIdx.x * blockDim.x + threadIdx.x;
  if (i >= K * N) return;
  int n = i / K, k = i % K;                 // wt is [N][K]
  wt[i] = f2bf(w[(size_t)k * N + n]);
}

__global__ void build_node_in(const float* __restrict__ x, const float* __restrict__ pos,
                              float* __restrict__ nin, int N) {
  int i = blockIdx.x * blockDim.x + threadIdx.x;
  if (i >= N) return;
  nin[i * 4 + 0] = x[i * 2 + 0];
  nin[i * 4 + 1] = x[i * 2 + 1];
  nin[i * 4 + 2] = pos[i * 2 + 0];
  nin[i * 4 + 3] = pos[i * 2 + 1];
}

// generic tiny linear (K<=8 or N<16), optional relu, f32 and/or bf16 output
__global__ void lin_small(const float* __restrict__ in, const float* __restrict__ w,
                          const float* __restrict__ b, float* __restrict__ outf,
                          unsigned short* __restrict__ outb, int M, int K, int N,
                          int relu) {
  int i = blockIdx.x * blockDim.x + threadIdx.x;
  if (i >= M * N) return;
  int m = i / N, n = i % N;
  float acc = b ? b[n] : 0.0f;
  for (int k = 0; k < K; ++k) acc += in[(size_t)m * K + k] * w[(size_t)k * N + n];
  if (relu && acc < 0.0f) acc = 0.0f;
  if (outf) outf[i] = acc;
  if (outb) outb[i] = f2bf(acc);
}

// per-edge physics terms + fused 5->32 / 6->32 relu linears (phys layer 0)
__global__ void physics_l0(const float* __restrict__ x, const float* __restrict__ pos,
                           const float* __restrict__ eattr, const int* __restrict__ src,
                           const int* __restrict__ dst, const float* __restrict__ wE,
                           const float* __restrict__ bE, const float* __restrict__ wH,
                           const float* __restrict__ bH, unsigned short* __restrict__ peb,
                           unsigned short* __restrict__ phb, int E) {
  int e = blockIdx.x * blockDim.x + threadIdx.x;
  if (e >= E) return;
  int s = src[e], d = dst[e];
  float dp0 = pos[d * 2] - pos[s * 2], dp1 = pos[d * 2 + 1] - pos[s * 2 + 1];
  float d2 = dp0 * dp0 + dp1 * dp1 + 1e-6f;
  float dE = x[d * 2] - x[s * 2];
  float g0 = dE * dp0 / d2, g1 = dE * dp1 / d2;
  float a0 = eattr[e * 4 + 0], a1 = eattr[e * 4 + 1];
  float a2 = eattr[e * 4 + 2], a3 = eattr[e * 4 + 3];
  float C = (a1 * dp0 - a0 * dp1) / d2;
  float pe[5] = {a0, a1, dp0, dp1, C};
  float ph[6] = {a0, a1, a2, a3, g0, g1};
  for (int n = 0; n < 32; ++n) {
    float u = bE[n];
    for (int k = 0; k < 5; ++k) u += pe[k] * wE[k * 32 + n];
    peb[(size_t)e * 32 + n] = f2bf(fmaxf(u, 0.0f));
    float v = bH[n];
    for (int k = 0; k < 6; ++k) v += ph[k] * wH[k * 32 + n];
    phb[(size_t)e * 32 + n] = f2bf(fmaxf(v, 0.0f));
  }
}

// out_bf16[i4] = bf16(a[i4] + bf16_to_f32(b[i4])), 4 elements per thread
__global__ void add_cast_bf(const float* __restrict__ a, const unsigned short* __restrict__ b,
                            unsigned short* __restrict__ out, long n4) {
  long i = (long)blockIdx.x * blockDim.x + threadIdx.x;
  if (i >= n4) return;
  float4 a4 = *(const float4*)(a + i * 4);
  uint2 b2 = *(const uint2*)(b + i * 4);
  uint2 o;
  o.x = (unsigned)f2bf(a4.x + bf2f((unsigned short)(b2.x & 0xFFFFu))) |
        ((unsigned)f2bf(a4.y + bf2f((unsigned short)(b2.x >> 16))) << 16);
  o.y = (unsigned)f2bf(a4.z + bf2f((unsigned short)(b2.y & 0xFFFFu))) |
        ((unsigned)f2bf(a4.w + bf2f((unsigned short)(b2.y >> 16))) << 16);
  *(uint2*)(out + i * 4) = o;
}

__global__ void fill_u32(unsigned* __restrict__ p, unsigned v, long n) {
  long i = (long)blockIdx.x * blockDim.x + threadIdx.x;
  if (i < n) p[i] = v;
}

// fo[e][c] += ani[src[e]][c] + anj[dst[e]][c]; 4 channels per thread (b128)
__global__ void combine_fo(float* __restrict__ fo, const float* __restrict__ ani,
                           const float* __restrict__ anj, const int* __restrict__ src,
                           const int* __restrict__ dst, int E) {
  int i = blockIdx.x * blockDim.x + threadIdx.x;
  if (i >= E * 32) return;
  int e = i >> 5, q = (i & 31) << 2;
  float4 f = *(float4*)(fo + (size_t)e * 128 + q);
  float4 a = *(const float4*)(ani + (size_t)src[e] * 128 + q);
  float4 b = *(const float4*)(anj + (size_t)dst[e] * 128 + q);
  f.x += a.x + b.x; f.y += a.y + b.y; f.z += a.z + b.z; f.w += a.w + b.w;
  *(float4*)(fo + (size_t)e * 128 + q) = f;
}

// a[e][h] = <leaky_relu(fo[e,h,:]), attn[h,:]> ; segment-max into mEnc[dst]
__global__ void attn_scores(const float* __restrict__ fo, const float* __restrict__ attnW,
                            const int* __restrict__ dst, float* __restrict__ a,
                            unsigned* __restrict__ mEnc, int E) {
  int i = blockIdx.x * blockDim.x + threadIdx.x;
  if (i >= E * 4) return;
  int e = i >> 2, h = i & 3;
  const float4* fr = (const float4*)(fo + (size_t)e * 128 + h * 32);
  const float4* wr = (const float4*)(attnW + h * 32);
  float s = 0.0f;
#pragma unroll
  for (int d = 0; d < 8; ++d) {
    float4 v = fr[d], w = wr[d];
    s += (v.x > 0.0f ? v.x : 0.2f * v.x) * w.x;
    s += (v.y > 0.0f ? v.y : 0.2f * v.y) * w.y;
    s += (v.z > 0.0f ? v.z : 0.2f * v.z) * w.z;
    s += (v.w > 0.0f ? v.w : 0.2f * v.w) * w.w;
  }
  a[i] = s;
  atomicMax(&mEnc[(size_t)dst[e] * 4 + h], fenc(s));
}

// ea = exp(a - m[dst]) in-place; segment-sum into ssum[dst]
__global__ void softmax_num(float* __restrict__ a, const unsigned* __restrict__ mEnc,
                            const int* __restrict__ dst, float* __restrict__ ssum, int E) {
  int i = blockIdx.x * blockDim.x + threadIdx.x;
  if (i >= E * 4) return;
  int e = i >> 2, h = i & 3;
  float m = fdec(mEnc[(size_t)dst[e] * 4 + h]);
  float ea = __expf(a[i] - m);
  a[i] = ea;
  atomicAdd(&ssum[(size_t)dst[e] * 4 + h], ea);
}

// nout[dst] += anode[src] * alpha ; 4 channels per thread (one b128 gather)
__global__ void scatter_msg(const float* __restrict__ ea, const float* __restrict__ ssum,
                            const float* __restrict__ anode, const int* __restrict__ src,
                            const int* __restrict__ dst, float* __restrict__ nout, int E) {
  int i = blockIdx.x * blockDim.x + threadIdx.x;
  if (i >= E * 32) return;
  int e = i >> 5, q = i & 31, h = q >> 3;   // c = 4q, h = c>>5
  int de = dst[e];
  float alpha = ea[(size_t)e * 4 + h] / (ssum[(size_t)de * 4 + h] + 1e-9f);
  float4 a = *(const float4*)(anode + (size_t)src[e] * 128 + (q << 2));
  float* np = nout + (size_t)de * 128 + (q << 2);
  atomicAdd(np + 0, a.x * alpha);
  atomicAdd(np + 1, a.y * alpha);
  atomicAdd(np + 2, a.z * alpha);
  atomicAdd(np + 3, a.w * alpha);
}

// h = LN(h + addend) (in-place f32) + packed bf16 copy; one wave per row of 128
__global__ void ln_residual(float* __restrict__ h, const float* __restrict__ addend,
                            const float* __restrict__ g, const float* __restrict__ b,
                            unsigned short* __restrict__ hb, int M) {
  int wave = threadIdx.x >> 5, lane = threadIdx.x & 31;
  int row = blockIdx.x * 8 + wave;
  if (row >= M) return;
  float* hr = h + (size_t)row * 128;
  float4 v = *(const float4*)(hr + lane * 4);
  if (addend) {
    float4 a = *(const float4*)(addend + (size_t)row * 128 + lane * 4);
    v.x += a.x; v.y += a.y; v.z += a.z; v.w += a.w;
  }
  float sum = v.x + v.y + v.z + v.w;
  for (int o = 16; o >= 1; o >>= 1) sum += __shfl_xor(sum, o, 32);
  float mu = sum * (1.0f / 128.0f);
  float dx = v.x - mu, dy = v.y - mu, dz = v.z - mu, dw = v.w - mu;
  float var = dx * dx + dy * dy + dz * dz + dw * dw;
  for (int o = 16; o >= 1; o >>= 1) var += __shfl_xor(var, o, 32);
  float inv = rsqrtf(var * (1.0f / 128.0f) + 1e-5f);
  float4 g4 = *(const float4*)(g + lane * 4);
  float4 b4 = *(const float4*)(b + lane * 4);
  float4 o4;
  o4.x = g4.x * dx * inv + b4.x;
  o4.y = g4.y * dy * inv + b4.y;
  o4.z = g4.z * dz * inv + b4.z;
  o4.w = g4.w * dw * inv + b4.w;
  *(float4*)(hr + lane * 4) = o4;
  uint2 p;
  p.x = (unsigned)f2bf(o4.x) | ((unsigned)f2bf(o4.y) << 16);
  p.y = (unsigned)f2bf(o4.z) | ((unsigned)f2bf(o4.w) << 16);
  *(uint2*)(hb + (size_t)row * 128 + lane * 4) = p;
}

// ---------------------------------------------------------------------------
// Host side
// ---------------------------------------------------------------------------
enum {
  IN_EDGE_ATTR = 0, IN_EDGE_INDEX = 1,
  P_EO_B0 = 2, P_EO_W0, P_EO_B1, P_EO_W1, P_EO_B2, P_EO_W2, P_EO_B3, P_EO_W3,
  P_GNN0 = 10, GNN_STRIDE = 9,
  GNN_ATTN = 0, GNN_FIJ_B = 1, GNN_FIJ_W = 2, GNN_NI_B = 3, GNN_NI_W = 4,
  GNN_NJ_B = 5, GNN_NJ_W = 6, GNN_NODE_B = 7, GNN_NODE_W = 8,
  P_INE_B0 = 46, P_INE_W0, P_INE_B1, P_INE_W1, P_INE_B2, P_INE_W2,
  P_INV_B0 = 52, P_INV_W0, P_INV_B1, P_INV_W1, P_INV_B2, P_INV_W2,
  P_LNE_B = 58, P_LNE_G = 59,
  P_LNEDGE0 = 60,  // (b,g) x 4
  P_LNNODE0 = 68,  // (b,g) x 4
  P_LNV_B = 76, P_LNV_G = 77,
  P_NO_B0 = 78, P_NO_W0, P_NO_B1, P_NO_W1, P_NO_B2, P_NO_W2, P_NO_B3, P_NO_W3,
  P_PE_B0 = 86, P_PE_W0, P_PE_B1, P_PE_W1,
  P_PH_B0 = 90, P_PH_W0, P_PH_B1, P_PH_W1,
  IN_POS = 94, IN_X = 95
};

static void launch_gemm(hipStream_t s, const unsigned short* A, const unsigned short* Wt,
                        const float* bias, float* Cf, unsigned short* Cb, int M, int K,
                        int N, bool relu) {
  dim3 grid(CEILDIV(M, 128)), block(256);
  if (K == 32 && N == 64)
    gemm_wmma<32, 64, 0, 1, true><<<grid, block, 0, s>>>(A, Wt, bias, Cf, Cb, M);
  else if (K == 64 && N == 128 && relu)
    gemm_wmma<64, 128, 1, 0, true><<<grid, block, 0, s>>>(A, Wt, bias, Cf, Cb, M);
  else if (K == 32 && N == 128)
    gemm_wmma<32, 128, 0, 1, false><<<grid, block, 0, s>>>(A, Wt, bias, Cf, Cb, M);
  else if (K == 128 && N == 128 && Cf)
    gemm_wmma<128, 128, 1, 0, false><<<grid, block, 0, s>>>(A, Wt, bias, Cf, Cb, M);
  else if (K == 128 && N == 128)
    gemm_wmma<128, 128, 0, 1, false><<<grid, block, 0, s>>>(A, Wt, bias, Cf, Cb, M);
  else if (K == 128 && N == 64)
    gemm_wmma<128, 64, 0, 1, false><<<grid, block, 0, s>>>(A, Wt, bias, Cf, Cb, M);
  else if (K == 64 && N == 32)
    gemm_wmma<64, 32, 1, 0, false><<<grid, block, 0, s>>>(A, Wt, bias, Cf, Cb, M);
}

extern "C" void kernel_launch(void* const* d_in, const int* in_sizes, int n_in,
                              void* d_out, int out_size, void* d_ws, size_t ws_size,
                              hipStream_t stream) {
  (void)n_in; (void)out_size; (void)ws_size;
  const int E = in_sizes[IN_EDGE_ATTR] / 4;
  const int N = in_sizes[IN_X] / 2;

  const float* edge_attr = (const float*)d_in[IN_EDGE_ATTR];
  const int* eidx = (const int*)d_in[IN_EDGE_INDEX];
  const int* src = eidx;
  const int* dst = eidx + E;
  const float* pos = (const float*)d_in[IN_POS];
  const float* x = (const float*)d_in[IN_X];
  auto P = [&](int i) { return (const float*)d_in[i]; };

  // bump allocator over d_ws
  char* base = (char*)d_ws;
  size_t off = 0;
  auto alloc = [&](size_t bytes) -> void* {
    void* p = base + off;
    off = (off + bytes + 255) & ~(size_t)255;
    return p;
  };

  // ---- pack weights: bf16, transposed to [N][K] ----
  auto pack = [&](int idx, int K, int Nn) -> unsigned short* {
    unsigned short* wt = (unsigned short*)alloc((size_t)K * Nn * 2);
    pack_wt<<<CEILDIV(K * Nn, 256), 256, 0, stream>>>(P(idx), wt, K, Nn);
    return wt;
  };
  unsigned short* Wt_inv1 = pack(P_INV_W1, 32, 64);
  unsigned short* Wt_inv2 = pack(P_INV_W2, 64, 128);
  unsigned short* Wt_ine1 = pack(P_INE_W1, 32, 64);
  unsigned short* Wt_ine2 = pack(P_INE_W2, 64, 128);
  unsigned short* Wt_pe1 = pack(P_PE_W1, 32, 128);
  unsigned short* Wt_ph1 = pack(P_PH_W1, 32, 128);
  unsigned short *Wt_fij[4], *Wt_ni[4], *Wt_nj[4], *Wt_node[4];
  for (int l = 0; l < 4; ++l) {
    int b = P_GNN0 + GNN_STRIDE * l;
    Wt_fij[l] = pack(b + GNN_FIJ_W, 128, 128);
    Wt_ni[l] = pack(b + GNN_NI_W, 128, 128);
    Wt_nj[l] = pack(b + GNN_NJ_W, 128, 128);
    Wt_node[l] = pack(b + GNN_NODE_W, 128, 128);
  }
  unsigned short* Wt_no0 = pack(P_NO_W0, 128, 128);
  unsigned short* Wt_no1 = pack(P_NO_W1, 128, 64);
  unsigned short* Wt_no2 = pack(P_NO_W2, 64, 32);
  unsigned short* Wt_eo0 = pack(P_EO_W0, 128, 128);
  unsigned short* Wt_eo1 = pack(P_EO_W1, 128, 64);
  unsigned short* Wt_eo2 = pack(P_EO_W2, 64, 32);

  // ---- activation buffers ----
  size_t N128 = (size_t)N * 128, E128 = (size_t)E * 128;
  float* hE_f = (float*)alloc(N128 * 4);
  unsigned short* hE_b = (unsigned short*)alloc(N128 * 2);
  unsigned short* nb0 = (unsigned short*)alloc(N128 * 2);
  unsigned short* nb1 = (unsigned short*)alloc(N128 * 2);
  float* ani = (float*)alloc(N128 * 4);
  float* anj = (float*)alloc(N128 * 4);
  float* anode = (float*)alloc(N128 * 4);
  float* nout = (float*)alloc(N128 * 4);
  unsigned* mEnc = (unsigned*)alloc((size_t)N * 4 * 4);
  float* ssum = (float*)alloc((size_t)N * 4 * 4);
  float* nin = (float*)alloc((size_t)N * 4 * 4);
  float* hH_f = (float*)alloc(E128 * 4);
  unsigned short* hH_b = (unsigned short*)alloc(E128 * 2);
  unsigned short* pE_b = (unsigned short*)alloc(E128 * 2);
  unsigned short* pH_b = (unsigned short*)alloc(E128 * 2);
  unsigned short* fin_b = (unsigned short*)alloc(E128 * 2);
  unsigned short* eb_t = (unsigned short*)alloc(E128 * 2);
  float* fo_f = (float*)alloc(E128 * 4);
  float* aea = (float*)alloc((size_t)E * 4 * 4);

  // ---- node encoder: [Ez,eps,pos] -> 32 -> 64 -> 128, then LN ----
  build_node_in<<<CEILDIV(N, 256), 256, 0, stream>>>(x, pos, nin, N);
  lin_small<<<CEILDIV(N * 32, 256), 256, 0, stream>>>(nin, P(P_INV_W0), P(P_INV_B0),
                                                      nullptr, nb0, N, 4, 32, 1);
  launch_gemm(stream, nb0, Wt_inv1, P(P_INV_B1), nullptr, nb1, N, 32, 64, true);
  launch_gemm(stream, nb1, Wt_inv2, P(P_INV_B2), hE_f, nullptr, N, 64, 128, true);
  ln_residual<<<CEILDIV(N, 8), 256, 0, stream>>>(hE_f, nullptr, P(P_LNV_G), P(P_LNV_B),
                                                 hE_b, N);

  // ---- edge encoder ----
  lin_small<<<CEILDIV(E * 32, 256), 256, 0, stream>>>(edge_attr, P(P_INE_W0),
                                                      P(P_INE_B0), nullptr, eb_t, E, 4,
                                                      32, 1);
  launch_gemm(stream, eb_t, Wt_ine1, P(P_INE_B1), nullptr, fin_b, E, 32, 64, true);
  launch_gemm(stream, fin_b, Wt_ine2, P(P_INE_B2), hH_f, nullptr, E, 64, 128, true);
  ln_residual<<<CEILDIV(E, 8), 256, 0, stream>>>(hH_f, nullptr, P(P_LNE_G), P(P_LNE_B),
                                                 hH_b, E);

  // ---- physics branches (layer0 fused scalar, layer1 WMMA) ----
  unsigned short* peb = eb_t;
  unsigned short* phb = eb_t + (size_t)E * 32;
  physics_l0<<<CEILDIV(E, 256), 256, 0, stream>>>(x, pos, edge_attr, src, dst,
                                                  P(P_PE_W0), P(P_PE_B0), P(P_PH_W0),
                                                  P(P_PH_B0), peb, phb, E);
  launch_gemm(stream, peb, Wt_pe1, P(P_PE_B1), nullptr, pE_b, E, 32, 128, false);
  launch_gemm(stream, phb, Wt_ph1, P(P_PH_B1), nullptr, pH_b, E, 32, 128, false);

  // ---- GNN layers (even: attention+node update; odd: edge update only) ----
  for (int l = 0; l < 4; ++l) {
    int b = P_GNN0 + GNN_STRIDE * l;
    bool even = (l % 2) == 0;
    launch_gemm(stream, hE_b, Wt_ni[l], P(b + GNN_NI_B), ani, nullptr, N, 128, 128, false);
    launch_gemm(stream, hE_b, Wt_nj[l], P(b + GNN_NJ_B), anj, nullptr, N, 128, 128, false);
    if (even)
      launch_gemm(stream, hE_b, Wt_node[l], P(b + GNN_NODE_B), anode, nullptr, N, 128,
                  128, false);
    add_cast_bf<<<CEILDIV(E * 32, 256), 256, 0, stream>>>(hH_f, even ? pE_b : pH_b,
                                                          fin_b, (long)E * 32);
    launch_gemm(stream, fin_b, Wt_fij[l], P(b + GNN_FIJ_B), fo_f, nullptr, E, 128, 128,
                false);
    combine_fo<<<CEILDIV(E * 32, 256), 256, 0, stream>>>(fo_f, ani, anj, src, dst, E);
    if (even) {
      fill_u32<<<CEILDIV(N * 4, 256), 256, 0, stream>>>(mEnc, 0x007FFFFFu, (long)N * 4);
      fill_u32<<<CEILDIV(N * 4, 256), 256, 0, stream>>>((unsigned*)ssum, 0u, (long)N * 4);
      fill_u32<<<CEILDIV(N * 128, 256), 256, 0, stream>>>((unsigned*)nout, 0u,
                                                          (long)N * 128);
      attn_scores<<<CEILDIV(E * 4, 256), 256, 0, stream>>>(fo_f, P(b + GNN_ATTN), dst,
                                                           aea, mEnc, E);
      softmax_num<<<CEILDIV(E * 4, 256), 256, 0, stream>>>(aea, mEnc, dst, ssum, E);
      scatter_msg<<<CEILDIV(E * 32, 256), 256, 0, stream>>>(aea, ssum, anode, src, dst,
                                                            nout, E);
      ln_residual<<<CEILDIV(N, 8), 256, 0, stream>>>(hE_f, nout, P(P_LNNODE0 + 2 * l + 1),
                                                     P(P_LNNODE0 + 2 * l), hE_b, N);
    } else {
      ln_residual<<<CEILDIV(E, 8), 256, 0, stream>>>(hH_f, fo_f, P(P_LNEDGE0 + 2 * l + 1),
                                                     P(P_LNEDGE0 + 2 * l), hH_b, E);
    }
  }

  // ---- output heads (no activations in reference) ----
  float* out_no = (float*)d_out;
  float* out_eo = out_no + N;
  launch_gemm(stream, hE_b, Wt_no0, P(P_NO_B0), nullptr, nb0, N, 128, 128, false);
  launch_gemm(stream, nb0, Wt_no1, P(P_NO_B1), nullptr, nb1, N, 128, 64, false);
  launch_gemm(stream, nb1, Wt_no2, P(P_NO_B2), ani, nullptr, N, 64, 32, false);
  lin_small<<<CEILDIV(N, 256), 256, 0, stream>>>(ani, P(P_NO_W3), P(P_NO_B3), out_no,
                                                 nullptr, N, 32, 1, 0);
  launch_gemm(stream, hH_b, Wt_eo0, P(P_EO_B0), nullptr, eb_t, E, 128, 128, false);
  launch_gemm(stream, eb_t, Wt_eo1, P(P_EO_B1), nullptr, fin_b, E, 128, 64, false);
  launch_gemm(stream, fin_b, Wt_eo2, P(P_EO_B2), fo_f, nullptr, E, 64, 32, false);
  lin_small<<<CEILDIV(E * 2, 256), 256, 0, stream>>>(fo_f, P(P_EO_W3), P(P_EO_B3),
                                                     out_eo, nullptr, E, 32, 2, 0);
}
